// GQAAttention_11441792877043
// MI455X (gfx1250) — matrix operand order repaired and checked by reading, not compiled, not run on hardware
//
#include <hip/hip_runtime.h>
#include <hip/hip_bf16.h>
#include <math.h>

// ---------------------------------------------------------------------------
// GQA attention for MI455X (gfx1250), wave32.
// bf16 WMMA for all matmuls; async global->LDS double buffering for both the
// GEMMs and the flash-attention K/V stream.
// ---------------------------------------------------------------------------

typedef __bf16  v16bf  __attribute__((ext_vector_type(16)));
typedef float   v8f    __attribute__((ext_vector_type(8)));
typedef unsigned short ushort8v __attribute__((ext_vector_type(8)));

#define S_LEN    2048
#define DMODEL   2048
#define NHEADS   32
#define NKV      8
#define HEADDIM  64
#define NBATCH   2
#define NROWS    (NBATCH * S_LEN)     // 4096 flattened (b, s) rows
#define KVDIM    (NKV * HEADDIM)      // 512

__device__ __forceinline__ unsigned short f32_to_bf16(float f) {
    unsigned int u = __float_as_uint(f);
    unsigned int r = u + 0x7FFFu + ((u >> 16) & 1u);   // round-to-nearest-even
    return (unsigned short)(r >> 16);
}

union frag16 { ushort8v us[2]; v16bf bf; };

// A-fragment (16x32 bf16, ISA 7.12.2): lanes 0-15 -> rows, K chunks {0..7,16..23};
// lanes 16-31 -> K chunks {8..15,24..31}.
__device__ __forceinline__ v16bf load_frag_a(const unsigned short* base, int ld,
                                             int row0, int k0) {
    int lane = threadIdx.x & 31;
    int idx  = lane & 15;
    int half = lane >> 4;
    const unsigned short* r = base + (size_t)(row0 + idx) * ld + k0;
    frag16 f;
    f.us[0] = *(const ushort8v*)(r + half * 8);
    f.us[1] = *(const ushort8v*)(r + 16 + half * 8);
    return f.bf;
}

// B-fragment (32x16 bf16): lane -> column N, K chunks match A-fragment
// (lanes 0-15: K {0..7,16..23}, lanes 16-31: K {8..15,24..31}).
__device__ __forceinline__ v16bf load_frag_b(const unsigned short* base, int ld,
                                             int col0, int k0) {
    int lane = threadIdx.x & 31;
    int idx  = lane & 15;
    int half = lane >> 4;
    const unsigned short* r = base + (size_t)(col0 + idx) * ld + k0;
    frag16 f;
    f.us[0] = *(const ushort8v*)(r + half * 8);
    f.us[1] = *(const ushort8v*)(r + 16 + half * 8);
    return f.bf;
}

__device__ __forceinline__ v8f wmma_bf16(v16bf a, v16bf b, v8f c) {
    return __builtin_amdgcn_wmma_f32_16x16x32_bf16(false, a, false, b,
                                                   (short)0, c, false, false);
}

__device__ __forceinline__ void async_b128(unsigned lds_addr, unsigned long long ga) {
    asm volatile("global_load_async_to_lds_b128 %0, %1, off"
                 :: "v"(lds_addr), "v"(ga) : "memory");
}

// Async copy of a 128-row x 64-col bf16 tile (16 KB): 1024 16B chunks,
// 4 unconditional async b128 issues per thread (no EXEC masking).
__device__ __forceinline__ void async_copy_tile64(const unsigned short* g, int ld,
                                                  int row0, int k0,
                                                  unsigned short* lds) {
#pragma unroll
    for (int i = 0; i < 4; ++i) {
        int c   = (int)threadIdx.x + i * 256;
        int row = c >> 3;
        int kc  = (c & 7) * 8;
        async_b128((unsigned)(size_t)(lds + row * 64 + kc),
                   (unsigned long long)(size_t)(g + (size_t)(row0 + row) * ld + k0 + kc));
    }
}

// ---------------------------------------------------------------------------
// Elementwise casts
// ---------------------------------------------------------------------------
__global__ void cast_f32_bf16(const float* __restrict__ in,
                              unsigned short* __restrict__ out, int n) {
    int i = blockIdx.x * blockDim.x + threadIdx.x;
    if (i < n) out[i] = f32_to_bf16(in[i]);
}

// ---------------------------------------------------------------------------
// GEMM: C[m,n] = sum_k A[m,k] * B[n,k]   (A (M,K) bf16, B (N,K) bf16, C f32)
// Block = 256 threads = 8 waves, 128x128 block tile, K-step 64 (2x32 WMMA
// substeps = 16 WMMAs per barrier).  A/B slabs double-buffered in LDS via
// async global->LDS loads:
//   top of iter: s_wait_asynccnt 0 (own writes of cur done) -> barrier
//   (publishes all waves' writes AND retires previous buffer reads) ->
//   issue next buffer's async loads -> compute from LDS.
// ---------------------------------------------------------------------------
__global__ __launch_bounds__(256)
void gemm_bf16_nt(const unsigned short* __restrict__ A,
                  const unsigned short* __restrict__ B,
                  float* __restrict__ C, int M, int N, int K) {
    __shared__ __align__(16) unsigned short As[2][128 * 64];   // 2 x 16 KB
    __shared__ __align__(16) unsigned short Bs[2][128 * 64];   // 2 x 16 KB

    int wave = threadIdx.x >> 5;
    int lane = threadIdx.x & 31;
    int mblk = blockIdx.y * 128;
    int nblk = blockIdx.x * 128;
    int wm   = (wave >> 1) * 32;
    int wn   = (wave & 1) * 64;

    v8f acc[2][4];
#pragma unroll
    for (int i = 0; i < 2; ++i)
#pragma unroll
        for (int j = 0; j < 4; ++j) acc[i][j] = (v8f)(0.0f);

    int nk = K >> 6;                         // 64-wide K slabs
    async_copy_tile64(A, K, mblk, 0, As[0]);
    async_copy_tile64(B, K, nblk, 0, Bs[0]);

    for (int kt = 0; kt < nk; ++kt) {
        int cur = kt & 1;
        asm volatile("s_wait_asynccnt 0x0" ::: "memory");
        __syncthreads();
        if (kt + 1 < nk) {
            async_copy_tile64(A, K, mblk, (kt + 1) << 6, As[cur ^ 1]);
            async_copy_tile64(B, K, nblk, (kt + 1) << 6, Bs[cur ^ 1]);
        }

        const unsigned short* as = As[cur];
        const unsigned short* bs = Bs[cur];
#pragma unroll
        for (int ks = 0; ks < 64; ks += 32) {
            v16bf a0 = load_frag_a(as, 64, wm,      ks);
            v16bf a1 = load_frag_a(as, 64, wm + 16, ks);
            v16bf b0 = load_frag_b(bs, 64, wn,      ks);
            v16bf b1 = load_frag_b(bs, 64, wn + 16, ks);
            v16bf b2 = load_frag_b(bs, 64, wn + 32, ks);
            v16bf b3 = load_frag_b(bs, 64, wn + 48, ks);

            acc[0][0] = wmma_bf16(a0, b0, acc[0][0]);
            acc[0][1] = wmma_bf16(a0, b1, acc[0][1]);
            acc[0][2] = wmma_bf16(a0, b2, acc[0][2]);
            acc[0][3] = wmma_bf16(a0, b3, acc[0][3]);
            acc[1][0] = wmma_bf16(a1, b0, acc[1][0]);
            acc[1][1] = wmma_bf16(a1, b1, acc[1][1]);
            acc[1][2] = wmma_bf16(a1, b2, acc[1][2]);
            acc[1][3] = wmma_bf16(a1, b3, acc[1][3]);
        }
    }

    // C/D layout: VGPR r -> row r (lanes 0-15) / r+8 (lanes 16-31), col = lane%16
    int m0  = mblk + wm;
    int n0  = nblk + wn;
    int col = lane & 15;
    int rb  = (lane >> 4) * 8;
#pragma unroll
    for (int i = 0; i < 2; ++i)
#pragma unroll
        for (int j = 0; j < 4; ++j)
#pragma unroll
            for (int r = 0; r < 8; ++r)
                C[(size_t)(m0 + i * 16 + rb + r) * N + n0 + j * 16 + col] =
                    acc[i][j][r];
}

// ---------------------------------------------------------------------------
// RoPE (LLaMA rotate-half) + f32 -> bf16; folds the 1/sqrt(64) scale into Q.
// ---------------------------------------------------------------------------
__global__ void rope_bf16(const float* __restrict__ in,
                          unsigned short* __restrict__ out,
                          int width, float scale) {
    int i = blockIdx.x * blockDim.x + threadIdx.x;
    int total = NROWS * width;
    if (i >= total) return;
    int row = i / width;               // b*S + s
    int col = i - row * width;
    int s   = row & (S_LEN - 1);
    int d   = col & (HEADDIM - 1);
    int j   = d & 31;
    // inv_freq = 10000^(-j/32) = exp(-j * ln(10000)/32)
    float inv = __expf(-0.28782313662425572f * (float)j);
    float ang = (float)s * inv;
    float sn, cs;
    __sincosf(ang, &sn, &cs);
    float v     = in[i];
    float other = (d < 32) ? -in[i + 32] : in[i - 32];
    out[i] = f32_to_bf16((v * cs + other * sn) * scale);
}

// ---------------------------------------------------------------------------
// V transpose: (b*S, kv*64+d) f32  ->  (b*512 + kv*64 + d, seq) bf16
// so the P@V B-operand reads contiguously along the key dimension.
// ---------------------------------------------------------------------------
__global__ void transpose_v_bf16(const float* __restrict__ Vf,
                                 unsigned short* __restrict__ Vt) {
    int i = blockIdx.x * blockDim.x + threadIdx.x;
    int total = NROWS * KVDIM;
    if (i >= total) return;
    int row = i / KVDIM;               // b*S + s
    int col = i - row * KVDIM;         // kv*64 + d
    int b   = row >> 11;
    int s   = row & (S_LEN - 1);
    Vt[(size_t)(b * KVDIM + col) * S_LEN + s] = f32_to_bf16(Vf[i]);
}

// ---------------------------------------------------------------------------
// Flash attention (no mask).  Block = 8 waves = 128 q-rows of one head;
// all waves share the head's K/V, so K (32x64) and V (64x32) tiles are
// async-staged into LDS once per 32-key step (double buffered, one barrier
// per step).  Per wave: S = Q@K^T via 4 WMMAs, online softmax with shfl_xor
// row reductions, P through private LDS (C-layout -> A-layout), P@V via
// 4 WMMAs.  Grid = B * NHEADS * (S/128) = 1024.
// ---------------------------------------------------------------------------
__global__ __launch_bounds__(256)
void attn_fa(const unsigned short* __restrict__ Qb,   // (4096, 2048) bf16
             const unsigned short* __restrict__ Kb,   // (4096, 512)  bf16
             const unsigned short* __restrict__ Vt,   // (1024, 2048) bf16
             float* __restrict__ Of) {                // (4096, 2048) f32
    __shared__ __align__(16) unsigned short Ks[2][32 * 64];   // 2 x 4 KB
    __shared__ __align__(16) unsigned short Vs[2][64 * 32];   // 2 x 4 KB
    __shared__ __align__(16) unsigned short pstage[8 * 512];  // 16x32 bf16/wave

    int wave = threadIdx.x >> 5;
    int lane = threadIdx.x & 31;
    int idx  = lane & 15;
    int half = lane >> 4;
    int tid  = threadIdx.x;

    int bid  = blockIdx.x;
    int qblk = bid & 15;                 // S/128 = 16
    int h    = (bid >> 4) & 31;
    int b    = bid >> 9;
    int kv   = h >> 2;                   // GQA: q-head h -> kv-head h/4
    int q0   = b * S_LEN + qblk * 128 + wave * 16;

    const unsigned short* Qh = Qb + h * HEADDIM;                       // ld DMODEL
    const unsigned short* Kh = Kb + (size_t)(b * S_LEN) * KVDIM + kv * HEADDIM;
    const unsigned short* Vh = Vt + (size_t)(b * KVDIM + kv * HEADDIM) * S_LEN;

    // per-thread async-copy coordinates (1 b128 chunk each for K and V tiles)
    int krow = tid >> 3, kcc = (tid & 7) * 8;   // K tile: 32 rows x 64 cols
    int vrow = tid >> 2, vcc = (tid & 3) * 8;   // V tile: 64 rows x 32 cols

    v16bf aq0 = load_frag_a(Qh, DMODEL, q0, 0);
    v16bf aq1 = load_frag_a(Qh, DMODEL, q0, 32);

    v8f o[4];
#pragma unroll
    for (int j = 0; j < 4; ++j) o[j] = (v8f)(0.0f);
    float m[8], l[8];
#pragma unroll
    for (int r = 0; r < 8; ++r) { m[r] = -3.0e30f; l[r] = 0.0f; }

    unsigned short* ps = pstage + wave * 512;

    // prime buffer 0 with the first 32-key tile
    async_b128((unsigned)(size_t)(Ks[0] + krow * 64 + kcc),
               (unsigned long long)(size_t)(Kh + (size_t)krow * KVDIM + kcc));
    async_b128((unsigned)(size_t)(Vs[0] + vrow * 32 + vcc),
               (unsigned long long)(size_t)(Vh + (size_t)vrow * S_LEN + vcc));

    for (int kt = 0; kt < S_LEN; kt += 32) {
        int cur = (kt >> 5) & 1;
        asm volatile("s_wait_asynccnt 0x0" ::: "memory");
        __syncthreads();
        if (kt + 32 < S_LEN) {
            async_b128((unsigned)(size_t)(Ks[cur ^ 1] + krow * 64 + kcc),
                       (unsigned long long)(size_t)(Kh + (size_t)(kt + 32 + krow) * KVDIM + kcc));
            async_b128((unsigned)(size_t)(Vs[cur ^ 1] + vrow * 32 + vcc),
                       (unsigned long long)(size_t)(Vh + (size_t)vrow * S_LEN + kt + 32 + vcc));
        }
        const unsigned short* ks = Ks[cur];
        const unsigned short* vs = Vs[cur];

        // ----- scores: two 16-key subtiles (scale already folded into Q) -----
        v8f s0 = (v8f)(0.0f), s1 = (v8f)(0.0f);
        s0 = wmma_bf16(aq0, load_frag_b(ks, 64, 0,  0),  s0);
        s0 = wmma_bf16(aq1, load_frag_b(ks, 64, 0,  32), s0);
        s1 = wmma_bf16(aq0, load_frag_b(ks, 64, 16, 0),  s1);
        s1 = wmma_bf16(aq1, load_frag_b(ks, 64, 16, 32), s1);

        // ----- online softmax; rows live in 16-lane halves -> xor 1,2,4,8 ----
#pragma unroll
        for (int r = 0; r < 8; ++r) {
            float red = fmaxf(s0[r], s1[r]);
            red = fmaxf(red, __shfl_xor(red, 1, 32));
            red = fmaxf(red, __shfl_xor(red, 2, 32));
            red = fmaxf(red, __shfl_xor(red, 4, 32));
            red = fmaxf(red, __shfl_xor(red, 8, 32));
            float mn   = fmaxf(m[r], red);
            float corr = __expf(m[r] - mn);
            float p0   = __expf(s0[r] - mn);
            float p1   = __expf(s1[r] - mn);
            float rs   = p0 + p1;
            rs += __shfl_xor(rs, 1, 32);
            rs += __shfl_xor(rs, 2, 32);
            rs += __shfl_xor(rs, 4, 32);
            rs += __shfl_xor(rs, 8, 32);
            l[r] = l[r] * corr + rs;
            m[r] = mn;
            o[0][r] *= corr; o[1][r] *= corr; o[2][r] *= corr; o[3][r] *= corr;
            int prow = r + half * 8;                 // C-layout row of VGPR r
            ps[prow * 32 + idx]      = f32_to_bf16(p0);
            ps[prow * 32 + 16 + idx] = f32_to_bf16(p1);
        }
        asm volatile("s_wait_dscnt 0x0" ::: "memory");

        // ----- reload P as an A-fragment (16x32) -----
        frag16 fp;
        const unsigned short* pr = ps + idx * 32;
        fp.us[0] = *(const ushort8v*)(pr + half * 8);
        fp.us[1] = *(const ushort8v*)(pr + 16 + half * 8);
        v16bf ap = fp.bf;

        // ----- o += P @ V (K=32 keys, N=64 head dims as 4 tiles) -----
        // (DS ops are in-order per wave: next iteration's pstage stores cannot
        //  pass this iteration's loads.)
#pragma unroll
        for (int j = 0; j < 4; ++j)
            o[j] = wmma_bf16(ap, load_frag_b(vs, 32, j * 16, 0), o[j]);
    }

    // ----- finalize: divide by row sums, store (b,s, h*64+d) f32 -----
#pragma unroll
    for (int r = 0; r < 8; ++r) {
        float invl = 1.0f / l[r];
        int row = q0 + r + half * 8;
#pragma unroll
        for (int j = 0; j < 4; ++j)
            Of[(size_t)row * DMODEL + h * HEADDIM + j * 16 + idx] = o[j][r] * invl;
    }
}

// ---------------------------------------------------------------------------
// Host launcher
// ---------------------------------------------------------------------------
extern "C" void kernel_launch(void* const* d_in, const int* in_sizes, int n_in,
                              void* d_out, int out_size, void* d_ws, size_t ws_size,
                              hipStream_t stream) {
    const float* x  = (const float*)d_in[0];
    const float* Wq = (const float*)d_in[1];
    const float* Wk = (const float*)d_in[2];
    const float* Wv = (const float*)d_in[3];
    const float* Wo = (const float*)d_in[4];
    float* out = (float*)d_out;

    char* ws = (char*)d_ws;
    size_t off = 0;
    auto alloc = [&](size_t bytes) -> void* {
        void* p = ws + off;
        off += (bytes + 255) & ~(size_t)255;
        return p;
    };

    unsigned short* xb  = (unsigned short*)alloc((size_t)NROWS * DMODEL * 2);
    unsigned short* Wqb = (unsigned short*)alloc((size_t)DMODEL * DMODEL * 2);
    unsigned short* Wkb = (unsigned short*)alloc((size_t)KVDIM * DMODEL * 2);
    unsigned short* Wvb = (unsigned short*)alloc((size_t)KVDIM * DMODEL * 2);
    unsigned short* Wob = (unsigned short*)alloc((size_t)DMODEL * DMODEL * 2);
    float* Qf = (float*)alloc((size_t)NROWS * DMODEL * 4);
    float* Kf = (float*)alloc((size_t)NROWS * KVDIM * 4);
    float* Vf = (float*)alloc((size_t)NROWS * KVDIM * 4);
    unsigned short* Qbq = (unsigned short*)alloc((size_t)NROWS * DMODEL * 2);
    unsigned short* Kbk = (unsigned short*)alloc((size_t)NROWS * KVDIM * 2);
    unsigned short* Vtb = (unsigned short*)alloc((size_t)NBATCH * KVDIM * S_LEN * 2);
    // Aliased scratch (dead-after-use buffers reused within this call):
    float*          Of = Qf;   // attention output reuses Qf
    unsigned short* Ob = xb;   // bf16 attention output reuses xb

    const int T = 256;
    int nXD = NROWS * DMODEL;   // 8.4M
    int nDD = DMODEL * DMODEL;  // 4.2M
    int nKD = KVDIM * DMODEL;   // 1.05M
    int nXK = NROWS * KVDIM;    // 2.1M

    cast_f32_bf16<<<(nXD + T - 1) / T, T, 0, stream>>>(x,  xb,  nXD);
    cast_f32_bf16<<<(nDD + T - 1) / T, T, 0, stream>>>(Wq, Wqb, nDD);
    cast_f32_bf16<<<(nKD + T - 1) / T, T, 0, stream>>>(Wk, Wkb, nKD);
    cast_f32_bf16<<<(nKD + T - 1) / T, T, 0, stream>>>(Wv, Wvb, nKD);
    cast_f32_bf16<<<(nDD + T - 1) / T, T, 0, stream>>>(Wo, Wob, nDD);

    dim3 gq(DMODEL / 128, NROWS / 128);   // (16, 32)
    dim3 gkv(KVDIM / 128, NROWS / 128);   // (4, 32)
    gemm_bf16_nt<<<gq,  T, 0, stream>>>(xb, Wqb, Qf, NROWS, DMODEL, DMODEL);
    gemm_bf16_nt<<<gkv, T, 0, stream>>>(xb, Wkb, Kf, NROWS, KVDIM,  DMODEL);
    gemm_bf16_nt<<<gkv, T, 0, stream>>>(xb, Wvb, Vf, NROWS, KVDIM,  DMODEL);

    rope_bf16<<<(nXD + T - 1) / T, T, 0, stream>>>(Qf, Qbq, DMODEL, 0.125f);
    rope_bf16<<<(nXK + T - 1) / T, T, 0, stream>>>(Kf, Kbk, KVDIM,  1.0f);
    transpose_v_bf16<<<(nXK + T - 1) / T, T, 0, stream>>>(Vf, Vtb);

    attn_fa<<<NBATCH * NHEADS * (S_LEN / 128), T, 0, stream>>>(Qbq, Kbk, Vtb, Of);

    cast_f32_bf16<<<(nXD + T - 1) / T, T, 0, stream>>>(Of, Ob, nXD);
    gemm_bf16_nt<<<gq, T, 0, stream>>>(Ob, Wob, out, NROWS, DMODEL, DMODEL);
}
